// Glow_76776835383759
// MI455X (gfx1250) — matrix operand order repaired and checked by reading, not compiled, not run on hardware
//
#include <hip/hip_runtime.h>
#include <hip/hip_bf16.h>
#include <cstdint>

// ---------------------------------------------------------------------------
// Problem constants (match reference)
// ---------------------------------------------------------------------------
#define BATCH 16384
#define DDIM  512
#define HDIM  1024
#define NLAY  8

typedef __attribute__((ext_vector_type(16))) __bf16 v16bf;
typedef __attribute__((ext_vector_type(8)))  float  v8f;

union FragB16 { uint32_t u[8]; v16bf v; };

#define ASYNC_WAIT() asm volatile("s_wait_asynccnt 0x0" ::: "memory")

// ---------------------------------------------------------------------------
// WMMA bf16 GEMM:  C[M,N] = A[M,K] * BT[N,K]^T   (A row-major, BT = B^T row-major)
// M = 16384, N in {512,1024}, K in {512,1024}. Tile: 128x128x32, 8 waves.
// Global -> LDS staging uses CDNA5 async-to-LDS DMA (ASYNCcnt), double buffered.
// Epilogue flag: 0 = linear (write f32 + masked bf16 z0)
//               1 = bias + relu -> bf16
//               2 = bias -> f32
// ---------------------------------------------------------------------------
#define BM 128
#define BN 128
#define BK 32
#define LDST 40   // padded LDS row stride in bf16 elements (80B, 16B aligned)

__global__ __launch_bounds__(256)
void k_gemm_bf16(const __bf16* __restrict__ A, const __bf16* __restrict__ BT,
                 int K, int N, float* __restrict__ outF, __bf16* __restrict__ outBF,
                 const float* __restrict__ bias, int flag, int lp)
{
    __shared__ unsigned short sA[2][BM * LDST];
    __shared__ unsigned short sB[2][BN * LDST];

    const int tid  = threadIdx.x;
    const int lane = tid & 31;
    const int w    = tid >> 5;
    const int wm   = w >> 2;   // 0..1  (M direction, 64 rows each)
    const int wn   = w & 3;    // 0..3  (N direction, 32 cols each)
    const int bm   = blockIdx.y * BM;
    const int bn   = blockIdx.x * BN;
    const int lh   = lane >> 4;   // lane half
    const int l15  = lane & 15;

    const __bf16* gA = A  + (size_t)bm * K;
    const __bf16* gB = BT + (size_t)bn * K;

    v8f acc[4][2] = {};

    const int KT = K / BK;

    // per-thread staging geometry: 2 chunks of 8 bf16 for A and for B
    const int r0 = tid >> 2;                 // chunk 0 row  (0..63)
    const int c0 = (tid & 3) * 8;            // chunk col
    const int r1 = r0 + 64;                  // chunk 1 row  (64..127)

    // issue async global->LDS DMA for K-tile kt into buffer buf
    auto issue = [&](int kt, int buf) {
        const __bf16* pA = gA + (size_t)kt * BK;
        const __bf16* pB = gB + (size_t)kt * BK;
        uint32_t la0 = (uint32_t)(uintptr_t)&sA[buf][r0 * LDST + c0];
        uint32_t la1 = (uint32_t)(uintptr_t)&sA[buf][r1 * LDST + c0];
        uint32_t lb0 = (uint32_t)(uintptr_t)&sB[buf][r0 * LDST + c0];
        uint32_t lb1 = (uint32_t)(uintptr_t)&sB[buf][r1 * LDST + c0];
        uint64_t ga0 = (uint64_t)(uintptr_t)(pA + (size_t)r0 * K + c0);
        uint64_t ga1 = (uint64_t)(uintptr_t)(pA + (size_t)r1 * K + c0);
        uint64_t gb0 = (uint64_t)(uintptr_t)(pB + (size_t)r0 * K + c0);
        uint64_t gb1 = (uint64_t)(uintptr_t)(pB + (size_t)r1 * K + c0);
        asm volatile("global_load_async_to_lds_b128 %0, %1, off"
                     :: "v"(la0), "v"(ga0) : "memory");
        asm volatile("global_load_async_to_lds_b128 %0, %1, off"
                     :: "v"(la1), "v"(ga1) : "memory");
        asm volatile("global_load_async_to_lds_b128 %0, %1, off"
                     :: "v"(lb0), "v"(gb0) : "memory");
        asm volatile("global_load_async_to_lds_b128 %0, %1, off"
                     :: "v"(lb1), "v"(gb1) : "memory");
    };

    issue(0, 0);
    ASYNC_WAIT();
    __syncthreads();

    for (int kt = 0; kt < KT; ++kt) {
        const int cur = kt & 1;
        if (kt + 1 < KT) issue(kt + 1, cur ^ 1);   // DMA next tile while computing

        // --- load fragments from LDS (ISA 7.12.2 16-bit layouts) ---
        FragB16 fa[4], fb[2];
#pragma unroll
        for (int mt = 0; mt < 4; ++mt) {
            int r = wm * 64 + mt * 16 + l15;
            const unsigned short* base = &sA[cur][r * LDST];
#pragma unroll
            for (int v = 0; v < 8; ++v) {
                int kk = ((v < 4) ? (v * 2) : (16 + (v - 4) * 2)) + 8 * lh; // A map
                fa[mt].u[v] = *(const uint32_t*)(base + kk);
            }
        }
#pragma unroll
        for (int nt = 0; nt < 2; ++nt) {
            int n = wn * 32 + nt * 16 + l15;
            const unsigned short* base = &sB[cur][n * LDST];
#pragma unroll
            for (int v = 0; v < 8; ++v) {
                int kk = ((v < 4) ? (v * 2) : (16 + (v - 4) * 2)) + 8 * lh; // B map
                fb[nt].u[v] = *(const uint32_t*)(base + kk);
            }
        }

#pragma unroll
        for (int mt = 0; mt < 4; ++mt)
#pragma unroll
            for (int nt = 0; nt < 2; ++nt)
                acc[mt][nt] = __builtin_amdgcn_wmma_f32_16x16x32_bf16(
                    false, fa[mt].v, false, fb[nt].v,
                    (short)0, acc[mt][nt], false, false);

        ASYNC_WAIT();       // my DMA writes for next tile are done
        __syncthreads();    // everyone done computing cur + all DMA visible
    }

    // --- epilogue: C/D layout: VGPR j -> row mt*16 + j + 8*lh, col nt*16 + l15
#pragma unroll
    for (int mt = 0; mt < 4; ++mt) {
#pragma unroll
        for (int nt = 0; nt < 2; ++nt) {
            int col = bn + wn * 32 + nt * 16 + l15;
#pragma unroll
            for (int j = 0; j < 8; ++j) {
                int row = bm + wm * 64 + mt * 16 + j + 8 * lh;
                float vv = acc[mt][nt][j];
                size_t idx = (size_t)row * N + col;
                if (flag == 0) {
                    outF[idx] = vv;
                    int mthis = ((col & 1) == lp);
                    outBF[idx] = (__bf16)(mthis ? vv : 0.0f);
                } else if (flag == 1) {
                    float t = vv + bias[col];
                    outBF[idx] = (__bf16)(t > 0.0f ? t : 0.0f);
                } else {
                    outF[idx] = vv + bias[col];
                }
            }
        }
    }
}

// ---------------------------------------------------------------------------
// Precompute kernels
// ---------------------------------------------------------------------------

// tmp = (L*tril(-1)+I) @ (U*triu(+1)+diag(sign*exp(log_s)))   [512x512 f32]
__global__ __launch_bounds__(256)
void k_lu_gemm(const float* __restrict__ Lm, const float* __restrict__ Um,
               const float* __restrict__ sign_s, const float* __restrict__ log_s,
               float* __restrict__ tmp, int l)
{
    __shared__ float sL[16][17];
    __shared__ float sU[16][17];
    const int tc = threadIdx.x, tr = threadIdx.y;
    const int r = blockIdx.y * 16 + tr;
    const int c = blockIdx.x * 16 + tc;
    const float* Lp = Lm + (size_t)l * DDIM * DDIM;
    const float* Up = Um + (size_t)l * DDIM * DDIM;
    const float diagc = sign_s[l * DDIM + c] * __expf(log_s[l * DDIM + c]);
    float accv = 0.0f;
    for (int k0 = 0; k0 < DDIM; k0 += 16) {
        int kL = k0 + tc;
        sL[tr][tc] = (kL < r) ? Lp[(size_t)r * DDIM + kL] : (kL == r ? 1.0f : 0.0f);
        int kU = k0 + tr;
        sU[tr][tc] = (kU < c) ? Up[(size_t)kU * DDIM + c] : (kU == c ? diagc : 0.0f);
        __syncthreads();
#pragma unroll
        for (int k = 0; k < 16; ++k) accv += sL[tr][k] * sU[k][tc];
        __syncthreads();
    }
    tmp[(size_t)r * DDIM + c] = accv;
}

// perm[r] = argmax_c P[l][r][c]  (P is a permutation matrix)
__global__ void k_perm(const float* __restrict__ P, int* __restrict__ perm, int l)
{
    int r = blockIdx.x * 256 + threadIdx.x;
    if (r >= DDIM) return;
    const float* row = P + ((size_t)l * DDIM + r) * DDIM;
    int idx = 0;
    for (int c = 0; c < DDIM; ++c) if (row[c] > 0.5f) idx = c;
    perm[r] = idx;
}

// Wbf[r][c] = bf16(tmp[perm[r]][c])  (row-major W == BT layout for z @ W^T)
__global__ void k_buildW(const float* __restrict__ tmp, const int* __restrict__ perm,
                         __bf16* __restrict__ Wbf)
{
    int i = blockIdx.x * 256 + threadIdx.x;      // 512*512 / 256 blocks
    int r = i >> 9, c = i & 511;
    Wbf[i] = (__bf16)tmp[(size_t)perm[r] * DDIM + c];
}

// out[C][R] = bf16(in[R][C])  — weight transpose+convert, 32x32 LDS tiles
__global__ __launch_bounds__(256)
void k_transpose(const float* __restrict__ in, __bf16* __restrict__ out, int R, int C)
{
    __shared__ float tile[32][33];
    const int bx = blockIdx.x * 32;   // C offset
    const int by = blockIdx.y * 32;   // R offset
    const int tx = threadIdx.x & 31;
    const int ty = threadIdx.x >> 5;  // 8 rows per pass
    for (int rr = ty; rr < 32; rr += 8)
        tile[rr][tx] = in[(size_t)(by + rr) * C + bx + tx];
    __syncthreads();
    for (int rr = ty; rr < 32; rr += 8)
        out[(size_t)(bx + rr) * R + by + tx] = (__bf16)tile[tx][rr];
}

// ---------------------------------------------------------------------------
// Per-layer elementwise kernels
// ---------------------------------------------------------------------------
__global__ void k_exp512(const float* __restrict__ ls, float* __restrict__ esc)
{
    int d = blockIdx.x * 256 + threadIdx.x;
    if (d < DDIM) esc[d] = __expf(ls[d]);
}

// actnorm: zabf = bf16(exp(log_scale[d]) * z + bias[d])
__global__ __launch_bounds__(256)
void k_actnorm(const float* __restrict__ zin, const float* __restrict__ esc,
               const float* __restrict__ bs, __bf16* __restrict__ out)
{
    size_t i = (size_t)blockIdx.x * 256 + threadIdx.x;
    int d = (int)(i & (DDIM - 1));
    out[i] = (__bf16)(esc[d] * zin[i] + bs[d]);
}

// coupling update + log-det row reduction (one block per batch row)
__global__ __launch_bounds__(256)
void k_update(const float* __restrict__ zl, const float* __restrict__ spre,
              const float* __restrict__ tpre, float* __restrict__ zout,
              const float* __restrict__ ldj_in, float* __restrict__ ldj_out, int lp)
{
    __shared__ float red[256];
    const int b = blockIdx.x;
    const int tid = threadIdx.x;
    const size_t base = (size_t)b * DDIM;
    float lsum = 0.0f;
    for (int d = tid; d < DDIM; d += 256) {
        float z = zl[base + d];
        float o;
        if ((d & 1) == lp) {            // m == 1: pass through
            o = z;
        } else {                        // m == 0: affine
            float s = tanhf(spre[base + d]);
            o = z * __expf(s) + tpre[base + d];
            lsum += s;
        }
        zout[base + d] = o;
    }
    red[tid] = lsum;
    __syncthreads();
    for (int off = 128; off > 0; off >>= 1) {
        if (tid < off) red[tid] += red[tid + off];
        __syncthreads();
    }
    if (tid == 0) ldj_out[b] = ldj_in[b] + red[0];
}

// constant ldj term:  sum over all layers of (sum log_scale + sum log_s)
__global__ void k_ldj_base(const float* __restrict__ log_scale,
                           const float* __restrict__ log_s, float* __restrict__ outScalar)
{
    __shared__ float red[256];
    const int tid = threadIdx.x;
    float s = 0.0f;
    for (int i = tid; i < NLAY * DDIM; i += 256) s += log_scale[i] + log_s[i];
    red[tid] = s;
    __syncthreads();
    for (int off = 128; off > 0; off >>= 1) {
        if (tid < off) red[tid] += red[tid + off];
        __syncthreads();
    }
    if (tid == 0) *outScalar = red[0];
}

__global__ void k_ldj_init(const float* __restrict__ scalar, float* __restrict__ ldj)
{
    int b = blockIdx.x * 256 + threadIdx.x;
    ldj[b] = *scalar;
}

// ---------------------------------------------------------------------------
// Host launcher
// ---------------------------------------------------------------------------
extern "C" void kernel_launch(void* const* d_in, const int* in_sizes, int n_in,
                              void* d_out, int out_size, void* d_ws, size_t ws_size,
                              hipStream_t stream)
{
    (void)in_sizes; (void)n_in; (void)out_size; (void)ws_size;
    const float* y         = (const float*)d_in[0];
    const float* log_scale = (const float*)d_in[1];
    const float* bias      = (const float*)d_in[2];
    const float* P         = (const float*)d_in[3];
    const float* Lm        = (const float*)d_in[4];
    const float* Um        = (const float*)d_in[5];
    const float* sign_s    = (const float*)d_in[6];
    const float* log_s     = (const float*)d_in[7];
    const float* Ws1       = (const float*)d_in[8];
    const float* bs1       = (const float*)d_in[9];
    const float* Ws2       = (const float*)d_in[10];
    const float* bs2       = (const float*)d_in[11];
    const float* Wt1       = (const float*)d_in[12];
    const float* bt1       = (const float*)d_in[13];
    const float* Wt2       = (const float*)d_in[14];
    const float* bt2       = (const float*)d_in[15];

    float* out_z   = (float*)d_out;
    float* out_ldj = out_z + (size_t)BATCH * DDIM;

    // ---- workspace carve ----
    char* w = (char*)d_ws;
    auto carve = [&](size_t bytes) -> void* {
        void* p = (void*)w;
        w += (bytes + 255) & ~(size_t)255;
        return p;
    };
    float*  zf32  = (float*) carve((size_t)BATCH * DDIM * 4);
    __bf16* zabf  = (__bf16*)carve((size_t)BATCH * DDIM * 2);
    __bf16* z0bf  = (__bf16*)carve((size_t)BATCH * DDIM * 2);
    __bf16* hsb   = (__bf16*)carve((size_t)BATCH * HDIM * 2);
    __bf16* htb   = (__bf16*)carve((size_t)BATCH * HDIM * 2);
    float*  spre  = (float*) carve((size_t)BATCH * DDIM * 4);
    float*  tpre  = (float*) carve((size_t)BATCH * DDIM * 4);
    float*  ldj   = (float*) carve((size_t)BATCH * 4);
    __bf16* Wbf   = (__bf16*)carve((size_t)DDIM * DDIM * 2);
    __bf16* Ws1T  = (__bf16*)carve((size_t)HDIM * DDIM * 2);
    __bf16* Wt1T  = (__bf16*)carve((size_t)HDIM * DDIM * 2);
    __bf16* Ws2T  = (__bf16*)carve((size_t)DDIM * HDIM * 2);
    __bf16* Wt2T  = (__bf16*)carve((size_t)DDIM * HDIM * 2);
    float*  tmp   = (float*) carve((size_t)DDIM * DDIM * 4);
    int*    perm  = (int*)   carve((size_t)DDIM * 4);
    float*  esc   = (float*) carve((size_t)DDIM * 4);
    float*  scal  = (float*) carve(256);

    // constant log-det contribution
    k_ldj_base<<<1, 256, 0, stream>>>(log_scale, log_s, scal);
    k_ldj_init<<<BATCH / 256, 256, 0, stream>>>(scal, ldj);

    const int anGrid = (BATCH * DDIM) / 256;

    for (int l = 0; l < NLAY; ++l) {
        const int lp = l & 1;
        const bool last = (l == NLAY - 1);

        // --- per-layer weight prep (cheap, L2-resident) ---
        k_lu_gemm<<<dim3(DDIM / 16, DDIM / 16), dim3(16, 16), 0, stream>>>(
            Lm, Um, sign_s, log_s, tmp, l);
        k_perm<<<2, 256, 0, stream>>>(P, perm, l);
        k_buildW<<<(DDIM * DDIM) / 256, 256, 0, stream>>>(tmp, perm, Wbf);
        k_transpose<<<dim3(HDIM / 32, DDIM / 32), 256, 0, stream>>>(
            Ws1 + (size_t)l * DDIM * HDIM, Ws1T, DDIM, HDIM);
        k_transpose<<<dim3(HDIM / 32, DDIM / 32), 256, 0, stream>>>(
            Wt1 + (size_t)l * DDIM * HDIM, Wt1T, DDIM, HDIM);
        k_transpose<<<dim3(DDIM / 32, HDIM / 32), 256, 0, stream>>>(
            Ws2 + (size_t)l * HDIM * DDIM, Ws2T, HDIM, DDIM);
        k_transpose<<<dim3(DDIM / 32, HDIM / 32), 256, 0, stream>>>(
            Wt2 + (size_t)l * HDIM * DDIM, Wt2T, HDIM, DDIM);

        // --- actnorm ---
        k_exp512<<<2, 256, 0, stream>>>(log_scale + (size_t)l * DDIM, esc);
        k_actnorm<<<anGrid, 256, 0, stream>>>(
            (l == 0) ? y : zf32, esc, bias + (size_t)l * DDIM, zabf);

        // --- invertible linear: zf32 = zabf @ W^T ; z0bf = m * zf32 (bf16) ---
        k_gemm_bf16<<<dim3(DDIM / BN, BATCH / BM), 256, 0, stream>>>(
            zabf, Wbf, DDIM, DDIM, zf32, z0bf, nullptr, 0, lp);

        // --- coupling MLP layer 1 (s and t paths), relu -> bf16 ---
        k_gemm_bf16<<<dim3(HDIM / BN, BATCH / BM), 256, 0, stream>>>(
            z0bf, Ws1T, DDIM, HDIM, nullptr, hsb, bs1 + (size_t)l * HDIM, 1, lp);
        k_gemm_bf16<<<dim3(HDIM / BN, BATCH / BM), 256, 0, stream>>>(
            z0bf, Wt1T, DDIM, HDIM, nullptr, htb, bt1 + (size_t)l * HDIM, 1, lp);

        // --- coupling MLP layer 2, bias -> f32 pre-activations ---
        k_gemm_bf16<<<dim3(DDIM / BN, BATCH / BM), 256, 0, stream>>>(
            hsb, Ws2T, HDIM, DDIM, spre, nullptr, bs2 + (size_t)l * DDIM, 2, lp);
        k_gemm_bf16<<<dim3(DDIM / BN, BATCH / BM), 256, 0, stream>>>(
            htb, Wt2T, HDIM, DDIM, tpre, nullptr, bt2 + (size_t)l * DDIM, 2, lp);

        // --- masked affine update + log-det reduction ---
        k_update<<<BATCH, 256, 0, stream>>>(
            zf32, spre, tpre,
            last ? out_z : zf32,
            ldj, last ? out_ldj : ldj, lp);
    }
}
